// ConvNetLayer_438086664819
// MI455X (gfx1250) — matrix operand order repaired and checked by reading, not compiled, not run on hardware
//
#include <hip/hip_runtime.h>

#define NN 16384
#define DD 64
#define KSPLIT 4
#define KSTEPS (NN / 32 / KSPLIT)     // 128 K-steps of 32 per wave

typedef __attribute__((ext_vector_type(16))) __bf16   v16bf;
typedef __attribute__((ext_vector_type(8)))  float    v8f;
typedef __attribute__((ext_vector_type(4)))  unsigned uint4v;

__device__ __forceinline__ unsigned short f32_to_bf16(float f) {
    unsigned u = __builtin_bit_cast(unsigned, f);
    unsigned r = u + 0x7FFFu + ((u >> 16) & 1u);   // round-to-nearest-even
    return (unsigned short)(r >> 16);
}
__device__ __forceinline__ unsigned pack2(unsigned short lo, unsigned short hi) {
    return (unsigned)lo | ((unsigned)hi << 16);
}

// ---------------------------------------------------------------------------
// Kernel 0: pre-pack x [N,64] f32 -> bf16 WMMA-B fragments.
// Fragment f = jblk*4 + dblk, 256 dwords each; lane l owns dwords
// [(f*32+l)*8, +8): element e has K=(l>>4)*16+e, N=l&15.
// ---------------------------------------------------------------------------
__global__ void __launch_bounds__(256) prepack_x_kernel(
        const float* __restrict__ x, unsigned* __restrict__ xp) {
    unsigned t    = blockIdx.x * 256u + threadIdx.x;
    unsigned dw   = t & 7u;
    unsigned lane = (t >> 3) & 31u;
    unsigned f    = t >> 8;
    unsigned dblk = f & 3u, jblk = f >> 2;
    unsigned hf   = lane >> 4, n = lane & 15u;
    unsigned j    = jblk * 32u + hf * 16u + 2u * dw;
    unsigned d    = dblk * 16u + n;
    float f0 = x[(size_t)j * DD + d];
    float f1 = x[(size_t)(j + 1) * DD + d];
    xp[t] = pack2(f32_to_bf16(f0), f32_to_bf16(f1));
}

// ---------------------------------------------------------------------------
// Stage-1 helpers (all fully unrolled so staging stays in VGPRs)
// ---------------------------------------------------------------------------
__device__ __forceinline__ void load_adj(const int* __restrict__ p, int hf,
                                         unsigned (&raw)[16]) {
    #pragma unroll
    for (int e = 0; e < 16; ++e) {
        int k = hf * 16 + e;   // A-element K
        raw[e] = (unsigned)__builtin_nontemporal_load(&p[k * NN]);
    }
}
__device__ __forceinline__ void load_x(const uint4v* __restrict__ xq, int lane,
                                       uint4v (&b)[8]) {
    #pragma unroll
    for (int db = 0; db < 4; ++db) {
        b[db * 2 + 0] = xq[(db * 32 + lane) * 2 + 0];
        b[db * 2 + 1] = xq[(db * 32 + lane) * 2 + 1];
    }
}
__device__ __forceinline__ void step_compute(const unsigned (&raw)[16],
                                             const uint4v (&b)[8],
                                             const v16bf& bones,
                                             v8f (&C)[4], v8f& Cdeg) {
    union { v16bf v; unsigned u[8]; } A;
    #pragma unroll
    for (int q = 0; q < 8; ++q)
        A.u[q] = raw[2 * q] * 0x3F80u + raw[2 * q + 1] * 0x3F800000u;
    #pragma unroll
    for (int db = 0; db < 4; ++db) {
        union { v16bf v; uint4v q2[2]; } B;
        B.q2[0] = b[db * 2 + 0];
        B.q2[1] = b[db * 2 + 1];
        C[db] = __builtin_amdgcn_wmma_f32_16x16x32_bf16(
            false, A.v, false, B.v, (short)0, C[db], false, false);
    }
    Cdeg = __builtin_amdgcn_wmma_f32_16x16x32_bf16(
        false, A.v, false, bones, (short)0, Cdeg, false, false);
}

// ---------------------------------------------------------------------------
// Kernel 1 (split-K): aggP[part] += adjT-tile x x-tile over one K quarter,
// degP[part] from a constant ones-column B fragment.  One block = one 16-row
// i-tile; its 4 waves each own one K quarter.  2-buffer software pipeline.
// ---------------------------------------------------------------------------
__global__ void __launch_bounds__(128) agg_kernel(
        const int* __restrict__ adj, const unsigned* __restrict__ xp,
        float* __restrict__ aggP, float* __restrict__ degP) {
    const int lane = threadIdx.x & 31;
    const int part = threadIdx.x >> 5;       // K quarter owned by this wave
    const int i0   = blockIdx.x * 16;
    const int hf   = lane >> 4;
    const int m    = lane & 15;              // A-matrix row of this lane
    const int kb0  = part * KSTEPS;

    const int*    p  = adj + (size_t)kb0 * 32 * NN + (i0 + m);
    const uint4v* xq = (const uint4v*)xp + (size_t)kb0 * 256;
    const size_t  pinc = (size_t)32 * NN;

    union { v16bf v; unsigned u[8]; } Bones;
    {
        unsigned od = (m == 0) ? 0x3F803F80u : 0u;   // ones in column N==0
        #pragma unroll
        for (int q = 0; q < 8; ++q) Bones.u[q] = od;
    }

    v8f C[4], Cdeg;
    v8f zero = {0.f,0.f,0.f,0.f,0.f,0.f,0.f,0.f};
    #pragma unroll
    for (int db = 0; db < 4; ++db) C[db] = zero;
    Cdeg = zero;

    unsigned A0[16], A1[16];
    uint4v   B0[8],  B1[8];

    // prologue: step 0 in flight
    load_adj(p, hf, A0); load_x(xq, lane, B0);
    p += pinc; xq += 256;

    for (int it = 0; it < KSTEPS / 2 - 1; ++it) {
        load_adj(p, hf, A1); load_x(xq, lane, B1);   // prefetch step 2it+1
        p += pinc; xq += 256;
        step_compute(A0, B0, Bones.v, C, Cdeg);      // compute step 2it
        load_adj(p, hf, A0); load_x(xq, lane, B0);   // prefetch step 2it+2
        p += pinc; xq += 256;
        step_compute(A1, B1, Bones.v, C, Cdeg);      // compute step 2it+1
    }
    // epilogue: last two steps
    load_adj(p, hf, A1); load_x(xq, lane, B1);
    step_compute(A0, B0, Bones.v, C, Cdeg);
    step_compute(A1, B1, Bones.v, C, Cdeg);

    // store partial tile + partial degree (plain stores -> deterministic)
    float* ag = aggP + (size_t)part * NN * DD;
    #pragma unroll
    for (int db = 0; db < 4; ++db) {
        #pragma unroll
        for (int r = 0; r < 8; ++r)
            ag[(size_t)(i0 + r + 8 * hf) * DD + db * 16 + m] = C[db][r];
    }
    if (m == 0) {
        float* dg = degP + (size_t)part * NN;
        #pragma unroll
        for (int r = 0; r < 8; ++r) dg[i0 + r + 8 * hf] = Cdeg[r];
    }
}

// ---------------------------------------------------------------------------
// Kernel 2: reduce the 4 K-partials, scale rows by 1/deg, multiply by U^T
// (WMMA bf16), relu, store.  out = relu(((sum_p aggP)/deg) @ U^T).
// ---------------------------------------------------------------------------
__global__ void __launch_bounds__(128) out_kernel(
        const float* __restrict__ aggP, const float* __restrict__ degP,
        const float* __restrict__ U, float* __restrict__ out) {
    const int lane = threadIdx.x & 31;
    const int wave = threadIdx.x >> 5;
    const int i0   = (blockIdx.x * 4 + wave) * 16;
    const int hf   = lane >> 4;
    const int m    = lane & 15;        // A-row; also B-fragment column n

    // degree of this lane's A-row (fixed summation order -> deterministic)
    float deg = 0.f;
    #pragma unroll
    for (int pp = 0; pp < KSPLIT; ++pp) deg += degP[(size_t)pp * NN + i0 + m];
    const float inv = 1.0f / deg;

    // B fragments: B[k,n] = U[n*64 + k]; per-lane elements contiguous in k
    union { v16bf v; unsigned u[8]; } B[2][4];
    #pragma unroll
    for (int kb = 0; kb < 2; ++kb) {
        #pragma unroll
        for (int db = 0; db < 4; ++db) {
            const float* up = U + (size_t)(db * 16 + m) * DD + kb * 32 + hf * 16;
            #pragma unroll
            for (int q = 0; q < 8; ++q)
                B[kb][db].u[q] = pack2(f32_to_bf16(up[2 * q]),
                                       f32_to_bf16(up[2 * q + 1]));
        }
    }

    v8f C[4];
    v8f zero = {0.f,0.f,0.f,0.f,0.f,0.f,0.f,0.f};
    #pragma unroll
    for (int db = 0; db < 4; ++db) C[db] = zero;

    #pragma unroll
    for (int kb = 0; kb < 2; ++kb) {
        const size_t base = (size_t)(i0 + m) * DD + kb * 32 + hf * 16;
        union { v16bf v; unsigned u[8]; } A;
        #pragma unroll
        for (int q = 0; q < 8; ++q) {
            int e0  = 2 * q;
            int off = e0;   // k minus hf*16
            float s0 = 0.f, s1 = 0.f;
            #pragma unroll
            for (int pp = 0; pp < KSPLIT; ++pp) {
                const float* ap = aggP + (size_t)pp * NN * DD + base;
                s0 += ap[off];
                s1 += ap[off + 1];
            }
            A.u[q] = pack2(f32_to_bf16(s0 * inv), f32_to_bf16(s1 * inv));
        }
        #pragma unroll
        for (int db = 0; db < 4; ++db)
            C[db] = __builtin_amdgcn_wmma_f32_16x16x32_bf16(
                false, A.v, false, B[kb][db].v, (short)0, C[db], false, false);
    }

    #pragma unroll
    for (int db = 0; db < 4; ++db) {
        #pragma unroll
        for (int r = 0; r < 8; ++r) {
            float v = C[db][r];
            out[(size_t)(i0 + r + 8 * hf) * DD + db * 16 + m] = v > 0.f ? v : 0.f;
        }
    }
}

// ---------------------------------------------------------------------------
extern "C" void kernel_launch(void* const* d_in, const int* in_sizes, int n_in,
                              void* d_out, int out_size, void* d_ws, size_t ws_size,
                              hipStream_t stream) {
    const float* x   = (const float*)d_in[0];   // [N, 64] f32
    const int*   adj = (const int*)d_in[1];     // [N, N] int32 {0,1}
    const float* U   = (const float*)d_in[2];   // [64, 64] f32
    float*       out = (float*)d_out;           // [N, 64] f32

    char* ws = (char*)d_ws;
    unsigned* xp   = (unsigned*)ws;                                   //  2 MB
    float*    aggP = (float*)(ws + (size_t)NN * DD * 2);              // 16 MB
    float*    degP = (float*)(ws + (size_t)NN * DD * 2
                                 + (size_t)KSPLIT * NN * DD * 4);     // 256 KB

    prepack_x_kernel<<<(NN * DD / 2) / 256, 256, 0, stream>>>(x, xp);
    agg_kernel<<<NN / 16, 128, 0, stream>>>(adj, xp, aggP, degP);
    out_kernel<<<NN / 64, 128, 0, stream>>>(aggP, degP, U, out);
}
